// ConvolutionalRecurrentBlocks_35553739276410
// MI455X (gfx1250) — matrix operand order repaired and checked by reading, not compiled, not run on hardware
//
#include <hip/hip_runtime.h>
#include <hip/hip_fp16.h>

typedef _Float16 v16h __attribute__((ext_vector_type(16)));
typedef float    v8f  __attribute__((ext_vector_type(8)));

#define SZ   128
#define HW   (SZ * SZ)
#define NB   4
#define NT   6
#define CH   64

// ---------------------------------------------------------------------------
// Weight repack: f32 [Co][cin][k][k] (u-gate = co 0..63 from srcA, r-gate =
// co 64..127 from srcB) -> f16 WMMA A-fragment layout:
//   dst[co_tile][kpos][chunk][lane][half],  lane = m + 16*(kk/16),
//   half = kk%16   (contiguous k per lane half, matches B loads, wave32)
// Channel remap: ci' < P1 -> (ci' < R1 ? ci' : zero), else R1 + (ci'-P1).
// One-time per launch; runtime divides are fine here.
// ---------------------------------------------------------------------------
__global__ void pack_weights(const float* __restrict__ srcA,
                             const float* __restrict__ srcB,
                             __half* __restrict__ dst,
                             int cototal, int ksq, int cipad, int cin_src,
                             int P1, int R1) {
  int n = cototal * ksq * cipad;
  int idx = blockIdx.x * blockDim.x + threadIdx.x;
  if (idx >= n) return;
  int cip  = idx % cipad;
  int rem  = idx / cipad;
  int kpos = rem % ksq;
  int co   = rem / ksq;
  int src_ci = (cip < P1) ? ((cip < R1) ? cip : -1) : (R1 + (cip - P1));
  if (src_ci >= cin_src) src_ci = -1;
  float v = 0.0f;
  if (src_ci >= 0) {
    const float* s = (co < 64) ? srcA : srcB;
    int c = co & 63;
    v = s[((size_t)c * cin_src + src_ci) * ksq + kpos];
  }
  int m = co & 15, cotile = co >> 4;
  int kk = cip & 31, chunk = cip >> 5;
  int h    = kk & 15;
  int lane = m + 16 * (kk >> 4);
  int nchunk = cipad >> 5;
  dst[((((size_t)cotile * ksq + kpos) * nchunk + chunk) * 32 + lane) * 16 + h] =
      (__half)v;
}

// ---------------------------------------------------------------------------
// NCHW f32 (optionally elementwise h*r) -> NHWC f16 (for LDS-friendly loads)
// ---------------------------------------------------------------------------
__global__ void nchw_to_nhwc16(const float* __restrict__ h,
                               const float* __restrict__ r,
                               __half* __restrict__ out16) {
  int idx = blockIdx.x * blockDim.x + threadIdx.x;   // over NB*HW
  if (idx >= NB * HW) return;
  int bb = idx >> 14, pix = idx & (HW - 1);
  const float* hb = h + (size_t)bb * CH * HW + pix;
  const float* rb = r ? (r + (size_t)bb * CH * HW + pix) : nullptr;
  __half* ob = out16 + (size_t)idx * CH;
  for (int c8 = 0; c8 < 8; ++c8) {
    union { __half hh[8]; uint4 u; } pk;
#pragma unroll
    for (int j = 0; j < 8; ++j) {
      int c = c8 * 8 + j;
      float v = hb[(size_t)c * HW];
      if (rb) v *= rb[(size_t)c * HW];
      pk.hh[j] = (__half)v;
    }
    *(uint4*)(ob + c8 * 8) = pk.u;
  }
}

// x f32 [B][T][3][H][W] -> f16 [T][B][H*W][16]  (3 real channels + 13 zeros)
__global__ void x_to_f16(const float* __restrict__ x, __half* __restrict__ x16) {
  int idx = blockIdx.x * blockDim.x + threadIdx.x;   // NT*NB*HW
  if (idx >= NT * NB * HW) return;
  int pix = idx & (HW - 1);
  int bt = idx >> 14;
  int t = bt >> 2, bb = bt & 3;
  union { __half hh[16]; uint4 u[2]; } pk;
#pragma unroll
  for (int c = 0; c < 16; ++c) pk.hh[c] = (__half)0.0f;
#pragma unroll
  for (int c = 0; c < 3; ++c)
    pk.hh[c] = (__half)x[(((size_t)bb * NT + t) * 3 + c) * HW + pix];
  *(uint4*)(x16 + (size_t)idx * 16)     = pk.u[0];
  *(uint4*)(x16 + (size_t)idx * 16 + 8) = pk.u[1];
}

// 1x1 projection conv (bandwidth-bound, fp32), writes d_out[0:24*64*HW]
__global__ void conv1x1(const float* __restrict__ x, const float* __restrict__ w,
                        const float* __restrict__ b, float* __restrict__ out) {
  int idx = blockIdx.x * blockDim.x + threadIdx.x;   // 24*HW
  if (idx >= NB * NT * HW) return;
  int n = idx >> 14, pix = idx & (HW - 1);
  float x0 = x[((size_t)n * 3 + 0) * HW + pix];
  float x1 = x[((size_t)n * 3 + 1) * HW + pix];
  float x2 = x[((size_t)n * 3 + 2) * HW + pix];
  for (int co = 0; co < CH; ++co) {
    out[((size_t)n * CH + co) * HW + pix] =
        b[co] + w[co * 3] * x0 + w[co * 3 + 1] * x1 + w[co * 3 + 2] * x2;
  }
}

// ---------------------------------------------------------------------------
// Implicit-GEMM conv via v_wmma_f32_16x16x32_f16, fully compile-time shaped.
// Block = one 32-pixel row strip of one (batch,row); each wave owns a 16-Co
// tile and two 16x16 accumulators. Full receptive field staged in LDS (NHWC,
// ci fastest). MODE 0: fused u/r gates (sigmoid) -> outU/outR (NCHW f32).
// MODE 1: o gate (tanh) + fused GRU update -> hState f32 + h16out NHWC f16.
// ---------------------------------------------------------------------------
template <int MODE, int K, int CIPAD, int C1PAD>
__global__ void conv_wmma(const __half* __restrict__ in1,
                          const __half* __restrict__ in2,
                          const __half* __restrict__ wpack,
                          const float* __restrict__ biasA,
                          const float* __restrict__ biasB,
                          float* __restrict__ outU, float* __restrict__ outR,
                          const float* __restrict__ uIn,
                          float* __restrict__ hState,
                          __half* __restrict__ h16out) {
  constexpr int PAD = K / 2;
  constexpr int WT = 31 + K;           // strip width + halo
  constexpr int CV = CIPAD >> 3;       // 8-half vectors per column
  constexpr int NCHUNK = CIPAD >> 5;
  constexpr int NWAVE = (MODE == 0) ? 8 : 4;

  __shared__ __align__(32) __half lds[K * WT * CIPAD];

  const int tid = threadIdx.x;
  const int bidx = blockIdx.x;
  const int sx = bidx & 3;
  const int y  = (bidx >> 2) & (SZ - 1);
  const int bb = bidx >> 9;
  const int x0 = sx * 32;

  // ---- stage receptive field into LDS: [dy][col][ci] (ci fastest) ----
  constexpr int TOTAL = K * WT * CV;
  for (int i = tid; i < TOTAL; i += 32 * NWAVE) {
    int civ = i % CV;
    int rem = i / CV;
    int col = rem % WT;
    int dy  = rem / WT;
    int row = y + dy - PAD;
    int gx  = x0 + col - PAD;
    uint4 v = {0u, 0u, 0u, 0u};
    if ((unsigned)row < (unsigned)SZ && (unsigned)gx < (unsigned)SZ) {
      int ci8 = civ << 3;
      if (ci8 < C1PAD)
        v = *(const uint4*)(in1 +
              ((((size_t)bb * SZ + row) * SZ + gx) * C1PAD + ci8));
      else if (ci8 < C1PAD + CH)
        v = *(const uint4*)(in2 +
              ((((size_t)bb * SZ + row) * SZ + gx) * CH + (ci8 - C1PAD)));
    }
    *(uint4*)(lds + ((size_t)i << 3)) = v;
  }
  __syncthreads();

  // ---- WMMA main loop (all strides compile-time) ----
  const int wv = tid >> 5;
  const int lane = tid & 31;
  const int lg = lane >> 4;
  const int ln = lane & 15;
  v8f acc0 = {0.f, 0.f, 0.f, 0.f, 0.f, 0.f, 0.f, 0.f};
  v8f acc1 = {0.f, 0.f, 0.f, 0.f, 0.f, 0.f, 0.f, 0.f};
  const v16h* wb =
      (const v16h*)wpack + (size_t)wv * (K * K) * NCHUNK * 32 + lane;
  const int cb = 16 * lg;              // lane-group K offset inside a 32-chunk
  const __half* lbase0 = lds + (size_t)ln * CIPAD + cb;
  const __half* lbase1 = lds + (size_t)(ln + 16) * CIPAD + cb;

  for (int dy = 0; dy < K; ++dy) {
#pragma unroll
    for (int dx = 0; dx < K; ++dx) {
      const int kpos = dy * K + dx;
      const int loff = (dy * WT + dx) * CIPAD;
#pragma unroll
      for (int ch = 0; ch < NCHUNK; ++ch) {
        v16h a = wb[(kpos * NCHUNK + ch) * 32];
        v16h b0 = *(const v16h*)(lbase0 + loff + ch * 32);
        v16h b1 = *(const v16h*)(lbase1 + loff + ch * 32);
        acc0 = __builtin_amdgcn_wmma_f32_16x16x32_f16(
            false, a, false, b0, (short)0, acc0, false, false);
        acc1 = __builtin_amdgcn_wmma_f32_16x16x32_f16(
            false, a, false, b1, (short)0, acc1, false, false);
      }
    }
  }

  // ---- epilogue ----
  const int co0 = wv * 16 + 8 * lg;    // 8 consecutive Co per lane (rr 0..7)
  if (MODE == 0) {
    const bool isU = (co0 < CH);       // uniform per wave
    const float* bias = isU ? biasA : biasB;
    float* op = isU ? outU : outR;
    const int cc0 = co0 & 63;
#pragma unroll
    for (int nt = 0; nt < 2; ++nt) {
      v8f& a = nt ? acc1 : acc0;
      int x = x0 + nt * 16 + ln;
#pragma unroll
      for (int rr = 0; rr < 8; ++rr) {
        float val = a[rr] + bias[cc0 + rr];
        val = 1.0f / (1.0f + __expf(-val));
        op[(((size_t)bb * CH + (cc0 + rr)) * SZ + y) * SZ + x] = val;
      }
    }
  } else {
#pragma unroll
    for (int nt = 0; nt < 2; ++nt) {
      v8f& a = nt ? acc1 : acc0;
      int x = x0 + nt * 16 + ln;
      union { __half hh[8]; uint4 u; } pk;
#pragma unroll
      for (int rr = 0; rr < 8; ++rr) {
        int co = co0 + rr;
        size_t idx = (((size_t)bb * CH + co) * SZ + y) * SZ + x;
        float o  = tanhf(a[rr] + biasA[co]);
        float u  = uIn[idx];
        float hp = hState[idx];
        float hn = hp * (1.0f - u) + o * u;
        hState[idx] = hn;
        pk.hh[rr] = (__half)hn;
      }
      *(uint4*)(h16out + ((((size_t)bb * SZ + y) * SZ + x) * CH + co0)) = pk.u;
    }
  }
}

// ---------------------------------------------------------------------------
extern "C" void kernel_launch(void* const* d_in, const int* in_sizes, int n_in,
                              void* d_out, int out_size, void* d_ws,
                              size_t ws_size, hipStream_t stream) {
  (void)in_sizes; (void)n_in; (void)out_size; (void)ws_size;
  const float* x    = (const float*)d_in[0];
  const float* w1x1 = (const float*)d_in[4];
  const float* b1x1 = (const float*)d_in[5];
  const float* wr[3], *br[3], *wu[3], *bu[3], *wo[3], *bo[3];
  for (int i = 0; i < 3; ++i) {
    wr[i] = (const float*)d_in[6 + 6 * i + 0];
    br[i] = (const float*)d_in[6 + 6 * i + 1];
    wu[i] = (const float*)d_in[6 + 6 * i + 2];
    bu[i] = (const float*)d_in[6 + 6 * i + 3];
    wo[i] = (const float*)d_in[6 + 6 * i + 4];
    bo[i] = (const float*)d_in[6 + 6 * i + 5];
  }

  // ---- carve workspace ----
  char* ws = (char*)d_ws;
  size_t off = 0;
  auto alloc = [&](size_t bytes) -> void* {
    void* p = ws + off;
    off += (bytes + 255) & ~(size_t)255;
    return p;
  };
  const size_t hBytes32 = (size_t)NB * CH * HW * 4;   // 16 MB
  const size_t hBytes16 = (size_t)NB * CH * HW * 2;   // 8 MB
  float* h32[3];  for (int i = 0; i < 3; ++i) h32[i] = (float*)alloc(hBytes32);
  float* u32 = (float*)alloc(hBytes32);
  float* r32 = (float*)alloc(hBytes32);
  __half* h16[3]; for (int i = 0; i < 3; ++i) h16[i] = (__half*)alloc(hBytes16);
  __half* hr16 = (__half*)alloc(hBytes16);
  __half* x16  = (__half*)alloc((size_t)NT * NB * HW * 16 * 2);
  const int ksz[3]    = {3, 5, 7};
  const int cipad_[3] = {96, 128, 128};
  const int cin_[3]   = {67, 128, 128};
  __half *pur[3], *po[3];
  for (int i = 0; i < 3; ++i) {
    int ksq = ksz[i] * ksz[i];
    pur[i] = (__half*)alloc((size_t)128 * ksq * cipad_[i] * 2);
    po[i]  = (__half*)alloc((size_t)64  * ksq * cipad_[i] * 2);
  }

  // ---- one-time prep (per launch) ----
  conv1x1<<<(NB * NT * HW + 255) / 256, 256, 0, stream>>>(x, w1x1, b1x1,
                                                          (float*)d_out);
  x_to_f16<<<(NT * NB * HW + 255) / 256, 256, 0, stream>>>(x, x16);
  for (int i = 0; i < 3; ++i) {
    hipMemcpyAsync(h32[i], d_in[1 + i], hBytes32, hipMemcpyDeviceToDevice,
                   stream);
    nchw_to_nhwc16<<<(NB * HW + 255) / 256, 256, 0, stream>>>(h32[i], nullptr,
                                                              h16[i]);
    int ksq = ksz[i] * ksz[i];
    int P1 = (i == 0) ? 16 : 64, R1 = (i == 0) ? 3 : 64;
    int nUR = 128 * ksq * cipad_[i];
    pack_weights<<<(nUR + 255) / 256, 256, 0, stream>>>(
        wu[i], wr[i], pur[i], 128, ksq, cipad_[i], cin_[i], P1, R1);
    int nO = 64 * ksq * cipad_[i];
    pack_weights<<<(nO + 255) / 256, 256, 0, stream>>>(
        wo[i], wo[i], po[i], 64, ksq, cipad_[i], cin_[i], P1, R1);
  }

  // ---- sequential ConvGRU time loop ----
  const int grid = NB * SZ * 4;   // (batch, row, 32-px strip)
  for (int t = 0; t < NT; ++t) {
    const __half* xin = x16 + (size_t)t * NB * HW * 16;

    // layer 0: k=3, cipad=96, c1pad=16 (x channels padded 3->16)
    conv_wmma<0, 3, 96, 16><<<grid, 256, 0, stream>>>(
        xin, h16[0], pur[0], bu[0], br[0], u32, r32, nullptr, nullptr, nullptr);
    nchw_to_nhwc16<<<(NB * HW + 255) / 256, 256, 0, stream>>>(h32[0], r32, hr16);
    conv_wmma<1, 3, 96, 16><<<grid, 128, 0, stream>>>(
        xin, hr16, po[0], bo[0], nullptr, nullptr, nullptr, u32, h32[0], h16[0]);

    // layer 1: k=5, cipad=128, c1pad=64 (input = h of layer 0)
    conv_wmma<0, 5, 128, 64><<<grid, 256, 0, stream>>>(
        h16[0], h16[1], pur[1], bu[1], br[1], u32, r32, nullptr, nullptr,
        nullptr);
    nchw_to_nhwc16<<<(NB * HW + 255) / 256, 256, 0, stream>>>(h32[1], r32, hr16);
    conv_wmma<1, 5, 128, 64><<<grid, 128, 0, stream>>>(
        h16[0], hr16, po[1], bo[1], nullptr, nullptr, nullptr, u32, h32[1],
        h16[1]);

    // layer 2: k=7, cipad=128, c1pad=64 (input = h of layer 1)
    conv_wmma<0, 7, 128, 64><<<grid, 256, 0, stream>>>(
        h16[1], h16[2], pur[2], bu[2], br[2], u32, r32, nullptr, nullptr,
        nullptr);
    nchw_to_nhwc16<<<(NB * HW + 255) / 256, 256, 0, stream>>>(h32[2], r32, hr16);
    conv_wmma<1, 7, 128, 64><<<grid, 128, 0, stream>>>(
        h16[1], hr16, po[2], bo[2], nullptr, nullptr, nullptr, u32, h32[2],
        h16[2]);
  }

  // ---- final hidden states -> d_out (after x0 block) ----
  float* outH = (float*)d_out + (size_t)NB * NT * CH * HW;
  for (int i = 0; i < 3; ++i)
    hipMemcpyAsync(outH + (size_t)i * NB * CH * HW, h32[i], hBytes32,
                   hipMemcpyDeviceToDevice, stream);
}
